// CausalSelfAttention_53652731461837
// MI455X (gfx1250) — matrix operand order repaired and checked by reading, not compiled, not run on hardware
//
#include <hip/hip_runtime.h>
#include <math.h>

// ---------------------------------------------------------------------------
// CausalSelfAttention forward for MI455X (gfx1250), wave32 + WMMA bf16.
// All GEMMs and attention matmuls run on v_wmma_f32_16x16x32_bf16 with f32
// accumulation. Inputs converted once to bf16 in workspace.
// GEMM staging uses GLOBAL_LOAD_ASYNC_TO_LDS (ASYNCcnt) when the toolchain
// exposes the builtin; otherwise falls back to register staging.
// ---------------------------------------------------------------------------

typedef __bf16 bf16;
typedef __attribute__((ext_vector_type(16))) __bf16 bf16x16;
typedef __attribute__((ext_vector_type(8)))  __bf16 bf16x8;
typedef __attribute__((ext_vector_type(4)))  __bf16 bf16x4;
typedef __attribute__((ext_vector_type(8)))  float  f32x8;

union Frag { bf16x16 v; bf16x8 h[2]; };

#define NB   4
#define NT   2048
#define NC   2048
#define NH   16
#define ND   128            // NC / NH
#define QK_SCALE 0.08838834764831843f   // 1/sqrt(128)

#if defined(__has_builtin)
#if __has_builtin(__builtin_amdgcn_global_load_async_to_lds_b128)
#define HAVE_ASYNC_LDS 1
#endif
#endif
#ifndef HAVE_ASYNC_LDS
#define HAVE_ASYNC_LDS 0
#endif

#if HAVE_ASYNC_LDS
#if __has_builtin(__builtin_amdgcn_s_wait_asynccnt)
#define WAIT_ASYNC() __builtin_amdgcn_s_wait_asynccnt(0)
#else
#define WAIT_ASYNC() asm volatile("s_wait_asynccnt 0" ::: "memory")
#endif
typedef int v4i __attribute__((__vector_size__(4 * sizeof(int))));
typedef __attribute__((address_space(1))) v4i* gas_v4i;
typedef __attribute__((address_space(3))) v4i* lds_v4i;
static __device__ __forceinline__ void async_cp16(const bf16* g, bf16* l) {
    // 16 bytes per lane, global -> LDS, tracked by ASYNCcnt.
    // Builtin prototype: (v4i AS1* src, v4i AS3* dst, imm offset, imm cpol)
    v4i* gg = (v4i*)const_cast<bf16*>(g);   // reinterpret (generic)
    v4i* ll = (v4i*)l;                      // reinterpret (generic)
    __builtin_amdgcn_global_load_async_to_lds_b128((gas_v4i)gg, (lds_v4i)ll, 0, 0);
}
#endif

static __device__ __forceinline__ f32x8 wmma_bf16(const Frag& a, const Frag& b, f32x8 c) {
    return __builtin_amdgcn_wmma_f32_16x16x32_bf16(
        /*neg_a=*/false, a.v, /*neg_b=*/false, b.v,
        /*c_mod=*/(short)0, c, /*reuse_a=*/false, /*reuse_b=*/false);
}

// ---------------------------------------------------------------------------
// Kernel 1: f32 -> bf16 convert (vectorized x4)
// ---------------------------------------------------------------------------
__global__ __launch_bounds__(256) void cvt_f32_bf16(const float* __restrict__ in,
                                                    bf16* __restrict__ out, int n) {
    int i = (blockIdx.x * 256 + threadIdx.x) * 4;
    if (i + 3 < n) {
        float4 f = *(const float4*)(in + i);
        bf16x4 o;
        o[0] = (bf16)f.x; o[1] = (bf16)f.y; o[2] = (bf16)f.z; o[3] = (bf16)f.w;
        *(bf16x4*)(out + i) = o;
    }
}

// ---------------------------------------------------------------------------
// Kernel 2/5: bf16 WMMA GEMM, C = A(MxK) * B(KxN), f32 accumulation.
// Block tile 128x128, BK=32, 8 waves (4x2), each wave 32x64 = 2x4 WMMA tiles.
// Double-buffered LDS; async global->LDS fills when available.
// ---------------------------------------------------------------------------
template <bool OUT_BF16>
__global__ __launch_bounds__(256) void gemm_wmma_bf16(
    const bf16* __restrict__ A, const bf16* __restrict__ Bm,
    void* __restrict__ Cout, int M, int N, int K) {
    constexpr int BK = 32, APAD = 8, BPAD = 8;
    __shared__ bf16 As[2][128][BK + APAD];
    __shared__ bf16 Bs[2][BK][128 + BPAD];

    const int tid  = threadIdx.x;
    const int lane = tid & 31;
    const int w    = tid >> 5;
    const int wr   = w >> 1;          // 0..3  (M direction, 32 rows each)
    const int wc   = w & 1;           // 0..1  (N direction, 64 cols each)
    const int g    = lane >> 4;       // half-wave group
    const int ml   = lane & 15;
    const int m0   = blockIdx.y * 128;
    const int n0   = blockIdx.x * 128;

    // global->LDS staging indices (256 threads, 2x16 bf16 each)
    const int ar = tid >> 1, ac = (tid & 1) * 16;   // A tile: 128x32
    const int br = tid >> 3, bc = (tid & 7) * 16;   // B tile: 32x128

    f32x8 acc[2][4] = {};

#if HAVE_ASYNC_LDS
    {   // prologue: async fill tile 0
        const bf16* ap = A + (size_t)(m0 + ar) * K + ac;
        const bf16* bp = Bm + (size_t)br * N + n0 + bc;
        async_cp16(ap,     &As[0][ar][ac]);
        async_cp16(ap + 8, &As[0][ar][ac + 8]);
        async_cp16(bp,     &Bs[0][br][bc]);
        async_cp16(bp + 8, &Bs[0][br][bc + 8]);
        WAIT_ASYNC();
    }
    __syncthreads();
#else
    bf16x8 ra0, ra1, rb0, rb1;
    {   // prologue: tile 0 via registers
        const bf16* ap = A + (size_t)(m0 + ar) * K + ac;
        ra0 = *(const bf16x8*)ap; ra1 = *(const bf16x8*)(ap + 8);
        const bf16* bp = Bm + (size_t)br * N + n0 + bc;
        rb0 = *(const bf16x8*)bp; rb1 = *(const bf16x8*)(bp + 8);
        *(bf16x8*)&As[0][ar][ac] = ra0; *(bf16x8*)&As[0][ar][ac + 8] = ra1;
        *(bf16x8*)&Bs[0][br][bc] = rb0; *(bf16x8*)&Bs[0][br][bc + 8] = rb1;
    }
    __syncthreads();
#endif

    const int ns = K / BK;
    int buf = 0;
    for (int s = 0; s < ns; ++s) {
#if HAVE_ASYNC_LDS
        if (s + 1 < ns) {   // async fill next tile into buf^1
            const bf16* ap = A + (size_t)(m0 + ar) * K + (s + 1) * BK + ac;
            const bf16* bp = Bm + (size_t)((s + 1) * BK + br) * N + n0 + bc;
            async_cp16(ap,     &As[buf ^ 1][ar][ac]);
            async_cp16(ap + 8, &As[buf ^ 1][ar][ac + 8]);
            async_cp16(bp,     &Bs[buf ^ 1][br][bc]);
            async_cp16(bp + 8, &Bs[buf ^ 1][br][bc + 8]);
        }
#else
        if (s + 1 < ns) {   // prefetch next tile into registers
            const bf16* ap = A + (size_t)(m0 + ar) * K + (s + 1) * BK + ac;
            ra0 = *(const bf16x8*)ap; ra1 = *(const bf16x8*)(ap + 8);
            const bf16* bp = Bm + (size_t)((s + 1) * BK + br) * N + n0 + bc;
            rb0 = *(const bf16x8*)bp; rb1 = *(const bf16x8*)(bp + 8);
        }
#endif
        if (s + 2 < ns) {   // cache prefetch hint two tiles ahead
            __builtin_prefetch(A + (size_t)(m0 + ar) * K + (s + 2) * BK + ac, 0, 0);
            __builtin_prefetch(Bm + (size_t)((s + 2) * BK + br) * N + n0 + bc, 0, 0);
        }
        // A fragments: lane(g,m): K chunks at cols 8g and 16+8g
        Frag af[2];
#pragma unroll
        for (int mi = 0; mi < 2; ++mi) {
            const bf16* row = &As[buf][wr * 32 + mi * 16 + ml][0];
            af[mi].h[0] = *(const bf16x8*)(row + 8 * g);
            af[mi].h[1] = *(const bf16x8*)(row + 16 + 8 * g);
        }
#pragma unroll
        for (int ni = 0; ni < 4; ++ni) {
            Frag bf;    // B fragment: lane(g,n): column n, K rows 8g+s and 16+8g+s
            const bf16* col = &Bs[buf][0][wc * 64 + ni * 16 + ml];
#pragma unroll
            for (int s2 = 0; s2 < 8; ++s2) {
                bf.v[s2]     = col[(8 * g + s2) * (128 + BPAD)];
                bf.v[8 + s2] = col[(16 + 8 * g + s2) * (128 + BPAD)];
            }
#pragma unroll
            for (int mi = 0; mi < 2; ++mi)
                acc[mi][ni] = wmma_bf16(af[mi], bf, acc[mi][ni]);
        }
#if HAVE_ASYNC_LDS
        if (s + 1 < ns) WAIT_ASYNC();
#else
        if (s + 1 < ns) {
            *(bf16x8*)&As[buf ^ 1][ar][ac] = ra0; *(bf16x8*)&As[buf ^ 1][ar][ac + 8] = ra1;
            *(bf16x8*)&Bs[buf ^ 1][br][bc] = rb0; *(bf16x8*)&Bs[buf ^ 1][br][bc + 8] = rb1;
        }
#endif
        __syncthreads();
        buf ^= 1;
    }

    // epilogue: C layout VGPR r: row = r + 8*g, col = ml
#pragma unroll
    for (int mi = 0; mi < 2; ++mi)
#pragma unroll
        for (int ni = 0; ni < 4; ++ni)
#pragma unroll
            for (int r = 0; r < 8; ++r) {
                const int row = m0 + wr * 32 + mi * 16 + r + 8 * g;
                const int col = n0 + wc * 64 + ni * 16 + ml;
                const float v = acc[mi][ni][r];
                if constexpr (OUT_BF16)
                    ((bf16*)Cout)[(size_t)row * N + col] = (bf16)v;
                else
                    ((float*)Cout)[(size_t)row * N + col] = v;
            }
}

// ---------------------------------------------------------------------------
// Kernel 3: RoPE + head split.
// qkv[B,T,3C] -> q[B,H,T,D], kT[B,H,D,T] (transposed for B-matrix rows),
//               v[B,H,T,D].  One thread = 8 consecutive d (4 rope pairs).
// ---------------------------------------------------------------------------
__global__ __launch_bounds__(256) void rope_split(
    const bf16* __restrict__ qkv, const float* __restrict__ cosp,
    const float* __restrict__ sinp, bf16* __restrict__ qh,
    bf16* __restrict__ kT, bf16* __restrict__ vh) {
    const int gid = blockIdx.x * 256 + threadIdx.x;
    const int t  = gid & (NT - 1);
    const int i8 = (gid >> 11) & 15;      // D/8 groups
    const int h  = (gid >> 15) & (NH - 1);
    const int b  = gid >> 19;
    const int d0 = i8 * 8;

    const size_t qoff = ((size_t)b * NT + t) * (3 * NC) + h * ND + d0;
    bf16x8 qv = *(const bf16x8*)(qkv + qoff);
    bf16x8 kv = *(const bf16x8*)(qkv + qoff + NC);
    bf16x8 vv = *(const bf16x8*)(qkv + qoff + 2 * NC);
    float4 cc = *(const float4*)(cosp + (size_t)t * (ND / 2) + d0 / 2);
    float4 ss = *(const float4*)(sinp + (size_t)t * (ND / 2) + d0 / 2);
    const float c[4] = {cc.x, cc.y, cc.z, cc.w};
    const float s[4] = {ss.x, ss.y, ss.z, ss.w};

    bf16x8 qo, ko;
#pragma unroll
    for (int i = 0; i < 4; ++i) {
        float q1 = (float)qv[2 * i], q2 = (float)qv[2 * i + 1];
        float k1 = (float)kv[2 * i], k2 = (float)kv[2 * i + 1];
        qo[2 * i]     = (bf16)(q1 * c[i] - q2 * s[i]);
        qo[2 * i + 1] = (bf16)(q1 * s[i] + q2 * c[i]);
        ko[2 * i]     = (bf16)(k1 * c[i] - k2 * s[i]);
        ko[2 * i + 1] = (bf16)(k1 * s[i] + k2 * c[i]);
    }
    const size_t bh = (size_t)b * NH + h;
    *(bf16x8*)(qh + (bh * NT + t) * ND + d0) = qo;
    *(bf16x8*)(vh + (bh * NT + t) * ND + d0) = vv;
#pragma unroll
    for (int i = 0; i < 8; ++i)
        kT[(bh * ND + d0 + i) * NT + t] = ko[i];
}

// ---------------------------------------------------------------------------
// Kernel 4: flash attention. One wave = 16 q rows; stream 32-key chunks with
// online softmax; S and PV via WMMA; P re-layout through per-wave LDS.
// All register-subscript loops fully unrolled (no v_movrels paths).
// ---------------------------------------------------------------------------
__global__ __launch_bounds__(256) void attn_flash(
    const bf16* __restrict__ qh, const bf16* __restrict__ kT,
    const bf16* __restrict__ vh, bf16* __restrict__ yh) {
    __shared__ bf16 Psh[8][16][40];   // per-wave 16x32 P tile (+pad)

    const int w    = threadIdx.x >> 5;
    const int lane = threadIdx.x & 31;
    const int g    = lane >> 4;
    const int ml   = lane & 15;

    const int qblk = blockIdx.x & (NT / 128 - 1);
    const int h    = (blockIdx.x / (NT / 128)) & (NH - 1);
    const int b    =  blockIdx.x / (NT / 128) / NH;
    const int q0   = qblk * 128 + w * 16;

    const size_t bh = (size_t)b * NH + h;
    const bf16* qb  = qh + bh * NT * ND;
    const bf16* ktb = kT + bh * ND * NT;
    const bf16* vb  = vh + bh * NT * ND;

    // Q fragments for all 4 K-chunks of D=128 (loaded once)
    Frag qf[4];
#pragma unroll
    for (int dc = 0; dc < 4; ++dc) {
        const bf16* row = qb + (size_t)(q0 + ml) * ND + dc * 32;
        qf[dc].h[0] = *(const bf16x8*)(row + 8 * g);
        qf[dc].h[1] = *(const bf16x8*)(row + 16 + 8 * g);
    }

    f32x8 O[8] = {};
    float mrun[8], lrun[8];
#pragma unroll
    for (int r = 0; r < 8; ++r) { mrun[r] = -3.0e38f; lrun[r] = 0.0f; }

    const int nch = (q0 + 16 + 31) / 32;
    for (int kc = 0; kc < nch; ++kc) {
        const int k0 = kc * 32;
        f32x8 S0 = {}, S1 = {};
#pragma unroll
        for (int dc = 0; dc < 4; ++dc) {
            Frag kb0, kb1;   // B-matrix: lane(g,n) = key n, d rows 8g+s and 16+8g+s
            const bf16* kr = ktb + (size_t)(dc * 32) * NT + k0 + ml;
#pragma unroll
            for (int s2 = 0; s2 < 8; ++s2) {
                kb0.v[s2]     = kr[(size_t)(8 * g + s2) * NT];
                kb0.v[8 + s2] = kr[(size_t)(16 + 8 * g + s2) * NT];
                kb1.v[s2]     = kr[(size_t)(8 * g + s2) * NT + 16];
                kb1.v[8 + s2] = kr[(size_t)(16 + 8 * g + s2) * NT + 16];
            }
            S0 = wmma_bf16(qf[dc], kb0, S0);
            S1 = wmma_bf16(qf[dc], kb1, S1);
        }
        // causal mask + online softmax (row = q0+r+8g spans lanes of the half)
#pragma unroll
        for (int r = 0; r < 8; ++r) {
            const int row = q0 + r + 8 * g;
            float s0 = S0[r] * QK_SCALE;
            float s1 = S1[r] * QK_SCALE;
            if (k0 + ml > row)      s0 = -1.0e30f;
            if (k0 + 16 + ml > row) s1 = -1.0e30f;
            float x = fmaxf(s0, s1);
#pragma unroll
            for (int msk = 1; msk < 16; msk <<= 1) x = fmaxf(x, __shfl_xor(x, msk, 32));
            const float mnew = fmaxf(mrun[r], x);
            const float p0 = __expf(s0 - mnew);
            const float p1 = __expf(s1 - mnew);
            float rs = p0 + p1;
#pragma unroll
            for (int msk = 1; msk < 16; msk <<= 1) rs += __shfl_xor(rs, msk, 32);
            const float alpha = __expf(mrun[r] - mnew);
            lrun[r] = lrun[r] * alpha + rs;
            mrun[r] = mnew;
#pragma unroll
            for (int ni = 0; ni < 8; ++ni) O[ni][r] *= alpha;
            Psh[w][r + 8 * g][ml]      = (bf16)p0;
            Psh[w][r + 8 * g][16 + ml] = (bf16)p1;
        }
        // reload P in A-matrix layout (wave-local LDS; backend inserts dscnt wait)
        Frag pa;
        {
            const bf16* prow = &Psh[w][ml][0];
            pa.h[0] = *(const bf16x8*)(prow + 8 * g);
            pa.h[1] = *(const bf16x8*)(prow + 16 + 8 * g);
        }
#pragma unroll
        for (int ni = 0; ni < 8; ++ni) {
            Frag vf;   // lane(g,n) = d column n, key rows k0+8g+s and k0+16+8g+s
            const bf16* vr = vb + (size_t)k0 * ND + ni * 16 + ml;
#pragma unroll
            for (int s2 = 0; s2 < 8; ++s2) {
                vf.v[s2]     = vr[(size_t)(8 * g + s2) * ND];
                vf.v[8 + s2] = vr[(size_t)(16 + 8 * g + s2) * ND];
            }
            O[ni] = wmma_bf16(pa, vf, O[ni]);
        }
    }

#pragma unroll
    for (int r = 0; r < 8; ++r) lrun[r] = 1.0f / lrun[r];
#pragma unroll
    for (int ni = 0; ni < 8; ++ni)
#pragma unroll
        for (int r = 0; r < 8; ++r) {
            const int row = q0 + r + 8 * g;
            const int col = h * ND + ni * 16 + ml;
            yh[((size_t)b * NT + row) * NC + col] = (bf16)(O[ni][r] * lrun[r]);
        }
}

// ---------------------------------------------------------------------------
// Launcher
// ---------------------------------------------------------------------------
extern "C" void kernel_launch(void* const* d_in, const int* in_sizes, int n_in,
                              void* d_out, int out_size, void* d_ws, size_t ws_size,
                              hipStream_t stream) {
    (void)in_sizes; (void)n_in; (void)out_size; (void)ws_size;
    const float* x    = (const float*)d_in[0];
    const float* cosp = (const float*)d_in[1];
    const float* sinp = (const float*)d_in[2];
    const float* Wqkv = (const float*)d_in[3];
    const float* Wout = (const float*)d_in[4];
    float* out = (float*)d_out;

    // workspace layout (bf16 buffers)
    char* p = (char*)d_ws;
    const size_t n_x    = (size_t)NB * NT * NC;       // 16.8M
    const size_t n_wqkv = (size_t)NC * 3 * NC;        // 12.6M
    const size_t n_wout = (size_t)NC * NC;            //  4.2M
    const size_t n_qkv  = (size_t)NB * NT * 3 * NC;   // 50.3M
    bf16* xb    = (bf16*)p;  p += n_x    * 2;
    bf16* wqkvb = (bf16*)p;  p += n_wqkv * 2;
    bf16* woutb = (bf16*)p;  p += n_wout * 2;
    bf16* qkvb  = (bf16*)p;  p += n_qkv  * 2;
    bf16* qhb   = (bf16*)p;  p += n_x    * 2;
    bf16* kTb   = (bf16*)p;  p += n_x    * 2;
    bf16* vhb   = (bf16*)p;  p += n_x    * 2;
    bf16* yhb   = qkvb;      // reuse qkv buffer after rope_split consumed it

    // 1) convert inputs to bf16
    cvt_f32_bf16<<<(int)(n_x    / 1024), 256, 0, stream>>>(x,    xb,    (int)n_x);
    cvt_f32_bf16<<<(int)(n_wqkv / 1024), 256, 0, stream>>>(Wqkv, wqkvb, (int)n_wqkv);
    cvt_f32_bf16<<<(int)(n_wout / 1024), 256, 0, stream>>>(Wout, woutb, (int)n_wout);

    // 2) qkv = x @ Wqkv   (M=8192, N=6144, K=2048)
    gemm_wmma_bf16<true><<<dim3(3 * NC / 128, NB * NT / 128), 256, 0, stream>>>(
        xb, wqkvb, qkvb, NB * NT, 3 * NC, NC);

    // 3) rope + split into heads (kT transposed)
    rope_split<<<(NB * NH * (ND / 8) * NT) / 256, 256, 0, stream>>>(
        qkvb, cosp, sinp, qhb, kTb, vhb);

    // 4) causal flash attention -> y [B,T,C] bf16
    attn_flash<<<NB * NH * (NT / 128), 256, 0, stream>>>(qhb, kTb, vhb, yhb);

    // 5) out = y @ Wout   (M=8192, N=2048, K=2048), f32 output
    gemm_wmma_bf16<false><<<dim3(NC / 128, NB * NT / 128), 256, 0, stream>>>(
        yhb, woutb, out, NB * NT, NC, NC);
}
